// OperatorRNN_75892072120675
// MI455X (gfx1250) — matrix operand order repaired and checked by reading, not compiled, not run on hardware
//
#include <hip/hip_runtime.h>
#include <hip/hip_bf16.h>
#include <math.h>

// Problem constants (from reference)
#define BS   256
#define IL   512
#define OL   64
#define H    1024
#define G    4096   // 4*H
#define OUTC 64

typedef __attribute__((ext_vector_type(8)))  __bf16 bf16x8;
typedef __attribute__((ext_vector_type(16))) __bf16 v16bf;
typedef __attribute__((ext_vector_type(8)))  float  v8f;
typedef __attribute__((ext_vector_type(4)))  float  f32x4;

static __device__ __forceinline__ v16bf concat8(bf16x8 lo, bf16x8 hi) {
    return __builtin_shufflevector(lo, hi, 0,1,2,3,4,5,6,7,8,9,10,11,12,13,14,15);
}
static __device__ __forceinline__ float sigm(float x) {
    return 1.0f / (1.0f + __expf(-x));
}

// ---------------------------------------------------------------------------
// Kernel 1: h0[b,h] = sum_i qf[b,i,h]*w_in[i] + b_in ; c0 = 0
// Pure HBM-bandwidth reduction over 512MB (~22us @ 23.3 TB/s).
// grid = BS*H/256 = 1024 blocks, 256 threads.
// ---------------------------------------------------------------------------
__global__ __launch_bounds__(256) void k_h0(const float* __restrict__ qf,
                                            const float* __restrict__ w_in,
                                            const float* __restrict__ b_in,
                                            __bf16* __restrict__ Hbf0,
                                            float* __restrict__ c_state) {
    const int b   = blockIdx.x >> 2;
    const int col = ((blockIdx.x & 3) << 8) + threadIdx.x;
    const float* p = qf + (size_t)b * IL * H + col;
    float acc = 0.0f;
#pragma unroll 8
    for (int i = 0; i < IL; ++i)
        acc += p[(size_t)i * H] * w_in[i];   // w_in[i] is wave-uniform -> s_load
    acc += b_in[0];
    const size_t idx = (size_t)b * H + col;
    Hbf0[idx]    = (__bf16)acc;
    c_state[idx] = 0.0f;
}

// ---------------------------------------------------------------------------
// Kernel 2: weight prep.
//   j < 4096 : Wc = bf16(w_ih+w_hh), Whh = bf16(w_hh),
//              bc[j] = b_ih[j]+b_hh[j],
//              g0vec[j] = start_token . w_ih[j,:] + bc[j]   (batch-invariant)
//   j >= 4096: bf16 copy of w_out rows.
// ---------------------------------------------------------------------------
__global__ __launch_bounds__(256) void k_prep(const float* __restrict__ w_ih,
                                              const float* __restrict__ w_hh,
                                              const float* __restrict__ b_ih,
                                              const float* __restrict__ b_hh,
                                              const float* __restrict__ start_tok,
                                              const float* __restrict__ w_out,
                                              __bf16* __restrict__ WcBf,
                                              __bf16* __restrict__ WhhBf,
                                              float* __restrict__ g0vec,
                                              float* __restrict__ bc,
                                              __bf16* __restrict__ WoutBf) {
    __shared__ float red[256];
    const int j   = blockIdx.x;
    const int tid = threadIdx.x;
    if (j < G) {
        float part = 0.0f;
        for (int k = tid; k < H; k += 256) {
            const size_t idx = (size_t)j * H + k;
            const float wi = w_ih[idx];
            const float wh = w_hh[idx];
            WcBf[idx]  = (__bf16)(wi + wh);
            WhhBf[idx] = (__bf16)wh;
            part += start_tok[k] * wi;
        }
        red[tid] = part;
        __syncthreads();
        for (int s = 128; s > 0; s >>= 1) {
            if (tid < s) red[tid] += red[tid + s];
            __syncthreads();
        }
        if (tid == 0) {
            const float bb = b_ih[j] + b_hh[j];
            bc[j]    = bb;
            g0vec[j] = red[0] + bb;
        }
    } else {
        const int o = j - G;
        for (int k = tid; k < H; k += 256)
            WoutBf[(size_t)o * H + k] = (__bf16)w_out[(size_t)o * H + k];
    }
}

// ---------------------------------------------------------------------------
// Kernel 3: one LSTM step, fused GEMM + gates.
//   gates[b, gcol] = Hin[b,:] . W[gcol,:]  (bf16 WMMA, f32 acc)  + gbias[gcol]
// Block = 16 batches x 64 h-cols (=> 4 gates x 64 = 256 gate cols).
// A tile staged in LDS via CDNA5 async global->LDS DMA (ASYNCcnt tracked);
// each wave accumulates its TWO 16x16 N-tiles inside the K loop so one
// LDS A-fragment feeds two WMMAs (halves ds traffic).
// grid = (BS/16)*(H/64) = 256 blocks, 256 threads.
// ---------------------------------------------------------------------------
__global__ __launch_bounds__(256, 2) void k_lstm_step(
        const __bf16* __restrict__ Hin,
        const __bf16* __restrict__ W,       // [G][H] row-major bf16
        const float*  __restrict__ gbias,   // [G]
        float*        __restrict__ c_state, // [BS][H]
        __bf16*       __restrict__ Hout,    // [BS][H]
        float*        __restrict__ hout) {  // d_out output region + t*H
    __shared__ __align__(16) __bf16 a_s[16 * H];   // 32 KB
    __shared__ float g_s[16 * 256];                // 16 KB

    const int tid = threadIdx.x;
    const int b0  = (blockIdx.x >> 4) << 4;   // batch tile base
    const int c0  = (blockIdx.x & 15) << 6;   // h-column base

    // Stage A (16 x 1024 bf16) into LDS with async DMA: 2048 x 16B chunks,
    // 8 per lane. Low 32 bits of a generic __shared__ pointer are the
    // workgroup-relative LDS offset (ISA 10.2).
#pragma unroll
    for (int it = 0; it < 8; ++it) {
        const int idx = tid + it * 256;
        const int m = idx >> 7, k8 = idx & 127;
        const unsigned lds_addr =
            (unsigned)(unsigned long long)(const void*)&a_s[m * H + k8 * 8];
        const __bf16* gptr = &Hin[(size_t)(b0 + m) * H + k8 * 8];
        asm volatile("global_load_async_to_lds_b128 %0, %1, off"
                     :: "v"(lds_addr), "v"(gptr) : "memory");
    }
    asm volatile("s_wait_asynccnt 0x0" ::: "memory");
    __syncthreads();

    const int wave  = tid >> 5;
    const int lane  = tid & 31;
    const int lm    = lane & 15;
    const int kbA   = (lane >> 4) << 3;  // A frag: +0 / +8 within 32-K chunk
    const int kbB   = (lane >> 4) << 3;  // B frag: +0 / +8 within 32-K chunk

    // This wave's two N-tiles (local tiles 2*wave, 2*wave+1).
    const int nt0   = wave * 2;
    const int nt1   = nt0 + 1;
    const int ncol0 = (nt0 >> 2) * H + c0 + ((nt0 & 3) << 4) + lm;
    const int ncol1 = (nt1 >> 2) * H + c0 + ((nt1 & 3) << 4) + lm;
    const __bf16* wrow0 = W + (size_t)ncol0 * H + kbB;
    const __bf16* wrow1 = W + (size_t)ncol1 * H + kbB;

    v8f acc0 = {0.f, 0.f, 0.f, 0.f, 0.f, 0.f, 0.f, 0.f};
    v8f acc1 = {0.f, 0.f, 0.f, 0.f, 0.f, 0.f, 0.f, 0.f};
#pragma unroll 4
    for (int kc = 0; kc < H; kc += 32) {
        bf16x8 alo = *(const bf16x8*)&a_s[lm * H + kc + kbA];
        bf16x8 ahi = *(const bf16x8*)&a_s[lm * H + kc + 16 + kbA];
        const v16bf afrag = concat8(alo, ahi);

        bf16x8 b0lo = *(const bf16x8*)(wrow0 + kc);
        bf16x8 b0hi = *(const bf16x8*)(wrow0 + kc + 16);
        const v16bf bfrag0 = concat8(b0lo, b0hi);
        bf16x8 b1lo = *(const bf16x8*)(wrow1 + kc);
        bf16x8 b1hi = *(const bf16x8*)(wrow1 + kc + 16);
        const v16bf bfrag1 = concat8(b1lo, b1hi);

        acc0 = __builtin_amdgcn_wmma_f32_16x16x32_bf16(
                   false, afrag, false, bfrag0, (short)0, acc0, false, false);
        acc1 = __builtin_amdgcn_wmma_f32_16x16x32_bf16(
                   false, afrag, false, bfrag1, (short)0, acc1, false, false);
    }

    const int rbase = (lane >> 4) << 3;
#pragma unroll
    for (int r = 0; r < 8; ++r) {
        g_s[(rbase + r) * 256 + nt0 * 16 + lm] = acc0[r];
        g_s[(rbase + r) * 256 + nt1 * 16 + lm] = acc1[r];
    }
    __syncthreads();

    // Fused LSTM elementwise over this block's 16 x 64 (batch, h) cells.
    for (int idx = tid; idx < 1024; idx += 256) {
        const int m  = idx >> 6, cc = idx & 63;
        const int b  = b0 + m;
        const int hc = c0 + cc;
        const float xi = g_s[m * 256 +       cc] + gbias[          hc];
        const float xf = g_s[m * 256 +  64 + cc] + gbias[    H +   hc];
        const float xg = g_s[m * 256 + 128 + cc] + gbias[2 * H +   hc];
        const float xo = g_s[m * 256 + 192 + cc] + gbias[3 * H +   hc];
        const size_t ci = (size_t)b * H + hc;
        const float c  = c_state[ci];
        const float cn = sigm(xf) * c + sigm(xi) * tanhf(xg);
        const float h  = sigm(xo) * tanhf(cn);
        c_state[ci] = cn;
        Hout[ci]    = (__bf16)h;
        hout[(size_t)b * (OL * H) + hc] = h;   // output[b, t, hc]
    }
}

// ---------------------------------------------------------------------------
// Kernel 4: probs = output @ w_out^T + b_out.  [16384 x 1024] x [1024 x 64].
// One wave per 16x16 tile; A converted f32->bf16 on the fly.
// grid = 512 blocks x 256 threads (= 4096 waves = 1024 M-tiles * 4 N-tiles).
// ---------------------------------------------------------------------------
__global__ __launch_bounds__(256) void k_proj(const float*  __restrict__ outF,
                                              const __bf16* __restrict__ WoutBf,
                                              const float*  __restrict__ b_out,
                                              float*        __restrict__ probs) {
    const int tid  = threadIdx.x;
    const int wv   = blockIdx.x * 8 + (tid >> 5);
    const int lane = tid & 31;
    const int mt    = wv >> 2;
    const int ntile = wv & 3;
    const int row0 = mt << 4;
    const int col0 = ntile << 4;
    const int lm   = lane & 15;
    const int kbA  = (lane >> 4) << 3;
    const int kbB  = (lane >> 4) << 3;

    const float*  arow = outF  + (size_t)(row0 + lm) * H + kbA;
    const __bf16* brow = WoutBf + (size_t)(col0 + lm) * H + kbB;

    v8f acc = {0.f, 0.f, 0.f, 0.f, 0.f, 0.f, 0.f, 0.f};
#pragma unroll 4
    for (int kc = 0; kc < H; kc += 32) {
        const f32x4 a0 = *(const f32x4*)(arow + kc);
        const f32x4 a1 = *(const f32x4*)(arow + kc + 4);
        const f32x4 a2 = *(const f32x4*)(arow + kc + 16);
        const f32x4 a3 = *(const f32x4*)(arow + kc + 20);
        v16bf afrag;
#pragma unroll
        for (int i = 0; i < 4; ++i) {
            afrag[i]      = (__bf16)a0[i];
            afrag[4 + i]  = (__bf16)a1[i];
            afrag[8 + i]  = (__bf16)a2[i];
            afrag[12 + i] = (__bf16)a3[i];
        }
        bf16x8 blo = *(const bf16x8*)(brow + kc);
        bf16x8 bhi = *(const bf16x8*)(brow + kc + 16);
        v16bf bfrag = concat8(blo, bhi);
        acc = __builtin_amdgcn_wmma_f32_16x16x32_bf16(
                  false, afrag, false, bfrag, (short)0, acc, false, false);
    }
    const float bias = b_out[col0 + lm];
    const int rbase = (lane >> 4) << 3;
#pragma unroll
    for (int r = 0; r < 8; ++r)
        probs[(size_t)(row0 + rbase + r) * OUTC + col0 + lm] = acc[r] + bias;
}

// ---------------------------------------------------------------------------
// Host launcher
// ---------------------------------------------------------------------------
extern "C" void kernel_launch(void* const* d_in, const int* in_sizes, int n_in,
                              void* d_out, int out_size, void* d_ws, size_t ws_size,
                              hipStream_t stream) {
    const float* qf        = (const float*)d_in[1];
    const float* start_tok = (const float*)d_in[4];
    const float* w_in      = (const float*)d_in[5];
    const float* b_in      = (const float*)d_in[6];
    const float* w_ih      = (const float*)d_in[7];
    const float* w_hh      = (const float*)d_in[8];
    const float* b_ih      = (const float*)d_in[9];
    const float* b_hh      = (const float*)d_in[10];
    const float* w_out     = (const float*)d_in[11];
    const float* b_out     = (const float*)d_in[12];

    float* out    = (float*)d_out;
    float* probs  = out;                                  // [BS,OL,OUTC]
    float* outputF = out + (size_t)BS * OL * OUTC;        // [BS,OL,H]

    // Workspace layout (~18.4 MB total)
    char* ws = (char*)d_ws;
    float*  c_state = (float*)(ws);                                 // 1 MB
    __bf16* Hbuf0   = (__bf16*)(ws + (1u << 20));                   // 512 KB
    __bf16* Hbuf1   = (__bf16*)(ws + (1u << 20) + (512u << 10));    // 512 KB
    __bf16* WcBf    = (__bf16*)(ws + (2u << 20));                   // 8 MB
    __bf16* WhhBf   = (__bf16*)(ws + (10u << 20));                  // 8 MB
    float*  g0vec   = (float*)(ws + (18u << 20));                   // 16 KB
    float*  bc      = (float*)(ws + (18u << 20) + (64u << 10));     // 16 KB
    __bf16* WoutBf  = (__bf16*)(ws + (18u << 20) + (128u << 10));   // 128 KB

    k_h0<<<dim3(BS * H / 256), dim3(256), 0, stream>>>(qf, w_in, b_in, Hbuf0, c_state);
    k_prep<<<dim3(G + OUTC), dim3(256), 0, stream>>>(w_ih, w_hh, b_ih, b_hh, start_tok,
                                                     w_out, WcBf, WhhBf, g0vec, bc, WoutBf);

    for (int t = 0; t < OL; ++t) {
        const __bf16* Hin  = (t & 1) ? Hbuf1 : Hbuf0;
        __bf16*       Hout = (t & 1) ? Hbuf0 : Hbuf1;
        k_lstm_step<<<dim3((BS / 16) * (H / 64)), dim3(256), 0, stream>>>(
            Hin,
            (t == 0) ? WhhBf : WcBf,
            (t == 0) ? g0vec : bc,
            c_state, Hout, outputF + (size_t)t * H);
    }

    k_proj<<<dim3(512), dim3(256), 0, stream>>>(outputF, WoutBf, b_out, probs);
}
